// Encoder_71648644432101
// MI455X (gfx1250) — compile-verified, matrix operand order checked by reading, not run on hardware
//
#include <hip/hip_runtime.h>
#include <cstdint>
#include <cstddef>

// ---------------- constants (match reference) ----------------
#define B_C      2
#define N_C      2048
#define DIM_C    512
#define HEADS_C  8
#define DHEAD_C  64
#define INNER_C  512
#define MEM_C    16
#define J_C      (N_C + MEM_C)        // 2064
#define JPAD_C   2176                 // 17*128: exact CTA-tile multiple AND multiple of 32
#define FF_C     (4 * DIM_C)          // 2048
#define DEPTH_C  2
#define TOPK_C   8
#define NREL_C   (JPAD_C + N_C)       // bucket table covers rel in [-(N-1), JPAD-1]

typedef __bf16 bf16_t;
typedef __attribute__((ext_vector_type(16))) __bf16 bf16x16;
typedef __attribute__((ext_vector_type(8)))  float  floatx8;

// ---------------- bf16 helpers (native converts -> single v_cvt) ----------------
__device__ __forceinline__ bf16_t f2bf(float f)  { return (bf16_t)f; }
__device__ __forceinline__ float  bf2f(bf16_t h) { return (float)h; }

// one 16B async global->LDS copy (CDNA5 GLOBAL_LOAD_ASYNC_TO_LDS_B128, GVS mode,
// ASYNCcnt-tracked; CDNA5 ISA)
__device__ __forceinline__ void async_copy16(unsigned lds_off, const bf16_t* base,
                                             unsigned byte_off) {
  asm volatile("global_load_async_to_lds_b128 %0, %1, %2"
               :: "v"(lds_off), "v"(byte_off), "s"(base) : "memory");
}
__device__ __forceinline__ void wait_async0() {
  asm volatile("s_wait_asynccnt 0x0" ::: "memory");
}

// ================= generic bf16 WMMA GEMM (A @ B^T) =================
// C[M,Nc] = A[M,K] (row-major, lda) @ B[Nc,K]^T (row-major, ldb) -- K contiguous in both.
// 256 threads = 8 waves; wave tile (MT*16)x64; CTA tile (WM*MT*16)x(WN*64); WM*WN==8.
// Double-buffered LDS, async-to-LDS staging, single barrier per K-step.
// ALL launches are exact tile multiples (no guards).
// EPI: 0 none | 1 +bias[col] | 2 +bias[col]+resid | 3 +bias, exact GELU | 4 +rel-pos bias
template<int WM, int WN, int MT, int EPI, bool OUTBF>
__global__ __launch_bounds__(256) void gemm_wmma(
    const bf16_t* __restrict__ A, const bf16_t* __restrict__ Bm,
    float* __restrict__ Cf, bf16_t* __restrict__ Cb,
    const float* __restrict__ bias, const float* __restrict__ resid,
    const int* __restrict__ btab, const float* __restrict__ remb,
    int M, int Nc, int K, int lda, int ldb, int ldc,
    long aStrideZ, long bStrideZ, long cStrideZ)
{
  constexpr int BM = WM * MT * 16;
  constexpr int BN = WN * 64;
  constexpr int SA = 40;   // padded LDS row stride in bf16 (80B: 16B-aligned, skews banks)
  constexpr int SB = 40;
  __shared__ __align__(16) bf16_t Atile[2 * BM * SA];
  __shared__ __align__(16) bf16_t Btile[2 * BN * SB];   // N-major: Btile[n][k]

  const int tid   = threadIdx.x;
  const int lane  = tid & 31;
  const int wave  = tid >> 5;
  const int waveM = wave % WM;
  const int waveN = wave / WM;
  const int mBase = blockIdx.y * BM;
  const int nBase = blockIdx.x * BN;
  const int z     = blockIdx.z;
  A  += (size_t)z * aStrideZ;
  Bm += (size_t)z * bStrideZ;

  floatx8 acc[MT][4] = {};

  // 16-bit A-matrix 16x32 fragment layout (ISA 7.12.2):
  // lane L(0..15): M=L, K in {0..7} (q0) and {16..23} (q1); lane L+16: K in {8..15},{24..31}
  const int mFrag = lane & 15;
  const int kHalf = (lane >> 4) * 8;

  union Frag { bf16x16 v; uint4 q[2]; };

  constexpr int AIT = (BM * 4) / 256;   // 16B transfers per thread for A tile
  constexpr int BIT = (BN * 4) / 256;   // 16B transfers per thread for B tile

  auto stage = [&](int k0, int buf) {
#pragma unroll
    for (int it = 0; it < AIT; ++it) {
      int idx = tid + it * 256;
      int r = idx >> 2, c = (idx & 3) * 8;
      unsigned lds = (unsigned)(size_t)(void*)&Atile[buf * BM * SA + r * SA + c];
      unsigned off = (unsigned)(((size_t)(mBase + r) * lda + k0 + c) * sizeof(bf16_t));
      async_copy16(lds, A, off);
    }
#pragma unroll
    for (int it = 0; it < BIT; ++it) {
      int idx = tid + it * 256;
      int r = idx >> 2, c = (idx & 3) * 8;
      unsigned lds = (unsigned)(size_t)(void*)&Btile[buf * BN * SB + r * SB + c];
      unsigned off = (unsigned)(((size_t)(nBase + r) * ldb + k0 + c) * sizeof(bf16_t));
      async_copy16(lds, Bm, off);
    }
  };

  stage(0, 0);
  int cur = 0;
  for (int k0 = 0; k0 < K; k0 += 32) {
    wait_async0();        // drain this wave's copies into buf[cur]
    __syncthreads();      // all waves' copies into buf[cur] are done; all reads of
                          // buf[cur^1] finished (they fed WMMAs before the barrier)
    if (k0 + 32 < K) stage(k0 + 32, cur ^ 1);

    // load ALL fragments, then issue the WMMA chain (loose dscnt waits)
    Frag a[MT];
#pragma unroll
    for (int m = 0; m < MT; ++m) {
      int aoff = cur * BM * SA + (waveM * MT * 16 + m * 16 + mFrag) * SA + kHalf;
      a[m].q[0] = *(const uint4*)&Atile[aoff];
      a[m].q[1] = *(const uint4*)&Atile[aoff + 16];
    }
    Frag b[4];
#pragma unroll
    for (int t = 0; t < 4; ++t) {
      int boff = cur * BN * SB + (waveN * 64 + t * 16 + mFrag) * SB + kHalf;
      b[t].q[0] = *(const uint4*)&Btile[boff];
      b[t].q[1] = *(const uint4*)&Btile[boff + 16];
    }
#pragma unroll
    for (int t = 0; t < 4; ++t)
#pragma unroll
      for (int m = 0; m < MT; ++m)
        acc[m][t] = __builtin_amdgcn_wmma_f32_16x16x32_bf16(
            false, a[m].v, false, b[t].v, (short)0, acc[m][t], false, false);
    cur ^= 1;
  }

  // ---- epilogue + store (C layout: VGPR r -> M=r (lanes 0-15) / M=8+r (lanes 16-31)) ----
  const int rowInTile = (lane >> 4) * 8;
  const int colInTile = lane & 15;
#pragma unroll
  for (int m = 0; m < MT; ++m) {
#pragma unroll
    for (int t = 0; t < 4; ++t) {
#pragma unroll
      for (int r = 0; r < 8; ++r) {
        int row = mBase + waveM * MT * 16 + m * 16 + rowInTile + r;
        int col = nBase + waveN * 64 + t * 16 + colInTile;
        float v = acc[m][t][r];
        if (EPI == 1) v += bias[col];
        else if (EPI == 2) v += bias[col] + resid[(size_t)row * ldc + col];
        else if (EPI == 3) { v += bias[col]; v = 0.5f * v * (1.0f + erff(v * 0.70710678118f)); }
        else if (EPI == 4) {
          int bk = btab[col - row + (N_C - 1)];
          v += remb[bk * HEADS_C + z];
        }
        size_t o = (size_t)row * ldc + col + (size_t)z * cStrideZ;
        if (OUTBF) Cb[o] = f2bf(v); else Cf[o] = v;
      }
    }
  }
}

// ================= layernorm (one wave per 512-wide row) =================
__global__ __launch_bounds__(256) void layernorm_k(
    const float* __restrict__ x, const float* __restrict__ g,
    const float* __restrict__ bsh, bf16_t* __restrict__ out, int rows)
{
  int wave = threadIdx.x >> 5, lane = threadIdx.x & 31;
  int row = blockIdx.x * 8 + wave;
  if (row >= rows) return;
  const float* xr = x + (size_t)row * DIM_C;
  float s = 0.f, sq = 0.f;
#pragma unroll
  for (int j = lane; j < DIM_C; j += 32) { float v = xr[j]; s += v; sq += v * v; }
#pragma unroll
  for (int o = 16; o > 0; o >>= 1) { s += __shfl_xor(s, o, 32); sq += __shfl_xor(sq, o, 32); }
  float mean = s * (1.0f / DIM_C);
  float var  = sq * (1.0f / DIM_C) - mean * mean;
  float inv  = rsqrtf(var + 1e-5f);
#pragma unroll
  for (int j = lane; j < DIM_C; j += 32) {
    float v = (xr[j] - mean) * inv * g[j] + bsh[j];
    out[(size_t)row * DIM_C + j] = f2bf(v);
  }
}

// ========== fused top-k(8) threshold + softmax, one wave per score row ==========
__global__ __launch_bounds__(256) void topk_softmax_k(
    const float* __restrict__ dots, bf16_t* __restrict__ attn, int rows)
{
  int wave = threadIdx.x >> 5, lane = threadIdx.x & 31;
  int row = blockIdx.x * 8 + wave;
  if (row >= rows) return;
  const float* src = dots + (size_t)row * JPAD_C;

  // per-lane sorted top-8 over strided chunk
  float r[TOPK_C];
#pragma unroll
  for (int t = 0; t < TOPK_C; ++t) r[t] = -3.4e38f;
  for (int j = lane; j < J_C; j += 32) {
    float v = src[j];
    if (v > r[TOPK_C - 1]) {
      r[TOPK_C - 1] = v;
#pragma unroll
      for (int t = TOPK_C - 1; t > 0; --t)
        if (r[t] > r[t - 1]) { float tmp = r[t]; r[t] = r[t - 1]; r[t - 1] = tmp; }
    }
  }
  // extract global top-8 by 8 rounds of wave arg-max over per-lane heads
  int p = 0; float thr = -3.4e38f, rowmax = -3.4e38f;
#pragma unroll
  for (int t = 0; t < TOPK_C; ++t) {
    float v = -3.4e38f;
#pragma unroll
    for (int q = 0; q < TOPK_C; ++q) if (p == q) v = r[q];
    int idx = lane;
#pragma unroll
    for (int o = 16; o > 0; o >>= 1) {
      float ov = __shfl_xor(v, o, 32);
      int   oi = __shfl_xor(idx, o, 32);
      if (ov > v || (ov == v && oi < idx)) { v = ov; idx = oi; }
    }
    if (t == 0) rowmax = v;
    thr = v;
    if (lane == idx) p++;
  }
  // masked softmax
  float s = 0.f;
  for (int j = lane; j < J_C; j += 32) {
    float x = src[j];
    if (x >= thr) s += __expf(x - rowmax);
  }
#pragma unroll
  for (int o = 16; o > 0; o >>= 1) s += __shfl_xor(s, o, 32);
  float invs = 1.0f / s;
  bf16_t* dst = attn + (size_t)row * JPAD_C;
  for (int j = lane; j < JPAD_C; j += 32) {
    float a = 0.f;
    if (j < J_C) { float x = src[j]; if (x >= thr) a = __expf(x - rowmax) * invs; }
    dst[j] = f2bf(a);
  }
}

// ================= small elementwise kernels =================
__global__ void bucket_k(int* __restrict__ tab) {
  int idx = blockIdx.x * blockDim.x + threadIdx.x;
  if (idx >= NREL_C) return;
  int rel = idx - (N_C - 1);
  int n = -rel;
  int ret = (n < 0) ? 16 : 0;
  n = n < 0 ? -n : n;
  int bucket;
  if (n < 8) bucket = n;
  else {
    float nf = (float)n;
    int vl = 8 + (int)(logf(nf * 0.125f) * (1.0f / logf(16.0f)) * 8.0f);
    if (vl > 15) vl = 15;
    bucket = vl;
  }
  tab[idx] = ret + bucket;
}

// transpose + convert: W[K][Nc] (f32) -> WT[Nc][K] (bf16), so GEMMs run A @ WT^T
__global__ void cvtT_k(const float* __restrict__ W, bf16_t* __restrict__ WT,
                       int K, int Ncols) {
  long idx = (long)blockIdx.x * blockDim.x + threadIdx.x;
  long total = (long)K * Ncols;
  if (idx >= total) return;
  int n = (int)(idx % Ncols);
  int k = (int)(idx / Ncols);
  WT[(long)n * K + k] = f2bf(W[idx]);
}

__global__ void copy_f32_k(const float* __restrict__ s, float* __restrict__ d, long n) {
  long i = (long)blockIdx.x * blockDim.x + threadIdx.x;
  if (i < n) d[i] = s[i];
}

// K rows MEM..MEM+N of kflat[b][j][c] from kv projection output
__global__ void build_kf_k(const bf16_t* __restrict__ kv, bf16_t* __restrict__ kf) {
  long idx = (long)blockIdx.x * blockDim.x + threadIdx.x;
  long total = (long)B_C * N_C * INNER_C;
  if (idx >= total) return;
  int c = (int)(idx % INNER_C);
  long t = idx / INNER_C;
  int i = (int)(t % N_C);
  int b = (int)(t / N_C);
  kf[((long)b * JPAD_C + MEM_C + i) * INNER_C + c] =
      kv[((long)b * N_C + i) * (2 * INNER_C) + c];
}

// kflat memory-slot rows [0,MEM) + zero pad rows [J,JPAD)
__global__ void build_kmem_k(const float* __restrict__ mk, bf16_t* __restrict__ kf) {
  const int ROWS = MEM_C + (JPAD_C - J_C);  // 128
  int idx = blockIdx.x * blockDim.x + threadIdx.x;
  int total = B_C * ROWS * INNER_C;
  if (idx >= total) return;
  int c = idx % INNER_C;
  int t = idx / INNER_C;
  int rr = t % ROWS;
  int b = t / ROWS;
  bf16_t v = f2bf(0.0f);
  int row;
  if (rr < MEM_C) {
    int h = c >> 6, d = c & 63;
    row = rr;
    v = f2bf(mk[((long)h * MEM_C + rr) * DHEAD_C + d]);
  } else {
    row = J_C + (rr - MEM_C);
  }
  kf[((long)b * JPAD_C + row) * INNER_C + c] = v;
}

// vT[b][h*64+d][j]: transposed V (mem slots + projected + zero pad), K-contiguous for GEMM
__global__ void build_vT_k(const bf16_t* __restrict__ kv, const float* __restrict__ mv,
                           bf16_t* __restrict__ vT) {
  long idx = (long)blockIdx.x * blockDim.x + threadIdx.x;
  long total = (long)B_C * INNER_C * JPAD_C;
  if (idx >= total) return;
  int j = (int)(idx % JPAD_C);
  long t = idx / JPAD_C;
  int c = (int)(t % INNER_C);
  int b = (int)(t / INNER_C);
  bf16_t v = f2bf(0.0f);
  if (j < MEM_C) {
    int h = c >> 6, d = c & 63;
    v = f2bf(mv[((long)h * MEM_C + j) * DHEAD_C + d]);
  } else if (j < J_C) {
    v = kv[((long)b * N_C + (j - MEM_C)) * (2 * INNER_C) + INNER_C + c];
  }
  vT[idx] = v;
}

// fold talking-heads + scale into the query: qmix[b,ko,i,h*64+d] = P[h,ko]*scale*q[b,i,h*64+d]
__global__ void build_qmix_k(const bf16_t* __restrict__ q, const float* __restrict__ P,
                             bf16_t* __restrict__ qm) {
  long idx = (long)blockIdx.x * blockDim.x + threadIdx.x;
  long total = (long)B_C * HEADS_C * N_C * INNER_C;
  if (idx >= total) return;
  int c = (int)(idx % INNER_C);
  long t = idx / INNER_C;
  int i  = (int)(t % N_C); t /= N_C;
  int ko = (int)(t % HEADS_C);
  int b  = (int)(t / HEADS_C);
  int h = c >> 6;
  float val = P[h * HEADS_C + ko] * 0.125f * bf2f(q[((long)b * N_C + i) * INNER_C + c]);
  qm[idx] = f2bf(val);
}

// ================= host orchestration =================
static inline dim3 grid1d(long n, int blk) { return dim3((unsigned)((n + blk - 1) / blk)); }

extern "C" void kernel_launch(void* const* d_in, const int* in_sizes, int n_in,
                              void* d_out, int out_size, void* d_ws, size_t ws_size,
                              hipStream_t stream) {
  (void)in_sizes; (void)n_in; (void)out_size; (void)ws_size;
  const float* x_in   = (const float*)d_in[0];
  const float* ln1_g  = (const float*)d_in[1];
  const float* ln1_b  = (const float*)d_in[2];
  const float* Wq     = (const float*)d_in[3];
  const float* Wkv    = (const float*)d_in[4];
  const float* Wo     = (const float*)d_in[5];
  const float* bo     = (const float*)d_in[6];
  const float* pproj  = (const float*)d_in[7];
  const float* mem_k  = (const float*)d_in[8];
  const float* mem_v  = (const float*)d_in[9];
  const float* ln2_g  = (const float*)d_in[10];
  const float* ln2_b  = (const float*)d_in[11];
  const float* W1     = (const float*)d_in[12];
  const float* b1     = (const float*)d_in[13];
  const float* W2     = (const float*)d_in[14];
  const float* b2     = (const float*)d_in[15];
  const float* remb   = (const float*)d_in[16];

  char* ws = (char*)d_ws;
  size_t off = 0;
  auto alloc = [&](size_t bytes) -> void* {
    void* p = ws + off;
    off += (bytes + 255) & ~(size_t)255;
    return p;
  };

  // workspace (~320 MB; dots/attn chunked per batch)
  float*  x_cur = (float*)  alloc((size_t)B_C * N_C * DIM_C * 4);
  bf16_t* h_bf  = (bf16_t*) alloc((size_t)B_C * N_C * DIM_C * 2);
  bf16_t* q_bf  = (bf16_t*) alloc((size_t)B_C * N_C * INNER_C * 2);
  bf16_t* kv_bf = (bf16_t*) alloc((size_t)B_C * N_C * 2 * INNER_C * 2);
  bf16_t* qmix  = (bf16_t*) alloc((size_t)B_C * HEADS_C * N_C * INNER_C * 2);
  bf16_t* kflat = (bf16_t*) alloc((size_t)B_C * JPAD_C * INNER_C * 2);
  bf16_t* vT    = (bf16_t*) alloc((size_t)B_C * INNER_C * JPAD_C * 2);
  float*  dots  = (float*)  alloc((size_t)HEADS_C * N_C * JPAD_C * 4);
  bf16_t* attn  = (bf16_t*) alloc((size_t)HEADS_C * N_C * JPAD_C * 2);
  bf16_t* aout  = (bf16_t*) alloc((size_t)B_C * N_C * INNER_C * 2);
  bf16_t* ffb   = (bf16_t*) alloc((size_t)B_C * N_C * FF_C * 2);
  bf16_t* wqT   = (bf16_t*) alloc((size_t)INNER_C * DIM_C * 2);
  bf16_t* wkvT  = (bf16_t*) alloc((size_t)2 * INNER_C * DIM_C * 2);
  bf16_t* woT   = (bf16_t*) alloc((size_t)DIM_C * INNER_C * 2);
  bf16_t* w1T   = (bf16_t*) alloc((size_t)FF_C * DIM_C * 2);
  bf16_t* w2T   = (bf16_t*) alloc((size_t)DIM_C * FF_C * 2);
  int*    btab  = (int*)    alloc((size_t)NREL_C * 4);

  const int BLK = 256;
  const long nTok = (long)B_C * N_C;                // 4096 rows
  bucket_k<<<grid1d(NREL_C, BLK), BLK, 0, stream>>>(btab);
  copy_f32_k<<<grid1d(nTok * DIM_C, BLK), BLK, 0, stream>>>(x_in, x_cur, nTok * DIM_C);

  for (int l = 0; l < DEPTH_C; ++l) {
    // weight convert + transpose to [Nc][K] bf16
    cvtT_k<<<grid1d((long)DIM_C * INNER_C, BLK), BLK, 0, stream>>>(
        Wq + (long)l * DIM_C * INNER_C, wqT, DIM_C, INNER_C);
    cvtT_k<<<grid1d((long)DIM_C * 2 * INNER_C, BLK), BLK, 0, stream>>>(
        Wkv + (long)l * DIM_C * 2 * INNER_C, wkvT, DIM_C, 2 * INNER_C);
    cvtT_k<<<grid1d((long)INNER_C * DIM_C, BLK), BLK, 0, stream>>>(
        Wo + (long)l * INNER_C * DIM_C, woT, INNER_C, DIM_C);
    cvtT_k<<<grid1d((long)DIM_C * FF_C, BLK), BLK, 0, stream>>>(
        W1 + (long)l * DIM_C * FF_C, w1T, DIM_C, FF_C);
    cvtT_k<<<grid1d((long)FF_C * DIM_C, BLK), BLK, 0, stream>>>(
        W2 + (long)l * FF_C * DIM_C, w2T, FF_C, DIM_C);

    // ---- attention ----
    layernorm_k<<<grid1d(nTok, 8), BLK, 0, stream>>>(x_cur, ln1_g + l * DIM_C,
                                                     ln1_b + l * DIM_C, h_bf, (int)nTok);
    gemm_wmma<4, 2, 2, 0, true><<<dim3(4, 32, 1), BLK, 0, stream>>>(
        h_bf, wqT, nullptr, q_bf, nullptr, nullptr, nullptr, nullptr,
        (int)nTok, INNER_C, DIM_C, DIM_C, DIM_C, INNER_C, 0, 0, 0);
    gemm_wmma<4, 2, 2, 0, true><<<dim3(8, 32, 1), BLK, 0, stream>>>(
        h_bf, wkvT, nullptr, kv_bf, nullptr, nullptr, nullptr, nullptr,
        (int)nTok, 2 * INNER_C, DIM_C, DIM_C, DIM_C, 2 * INNER_C, 0, 0, 0);

    build_kmem_k<<<grid1d((long)B_C * 128 * INNER_C, BLK), BLK, 0, stream>>>(
        mem_k + (long)l * HEADS_C * MEM_C * DHEAD_C, kflat);
    build_kf_k<<<grid1d((long)B_C * N_C * INNER_C, BLK), BLK, 0, stream>>>(kv_bf, kflat);
    build_vT_k<<<grid1d((long)B_C * INNER_C * JPAD_C, BLK), BLK, 0, stream>>>(
        kv_bf, mem_v + (long)l * HEADS_C * MEM_C * DHEAD_C, vT);
    build_qmix_k<<<grid1d((long)B_C * HEADS_C * N_C * INNER_C, BLK), BLK, 0, stream>>>(
        q_bf, pproj + (long)l * HEADS_C * HEADS_C, qmix);

    for (int b = 0; b < B_C; ++b) {
      // scores: per output head z, dots[z,i,j] = qmix[b,z] @ kflat[b]^T + relbias
      gemm_wmma<4, 2, 2, 4, false><<<dim3(JPAD_C / 128, N_C / 128, HEADS_C),
                                     BLK, 0, stream>>>(
          qmix + (size_t)b * HEADS_C * N_C * INNER_C,
          kflat + (size_t)b * JPAD_C * INNER_C,
          dots, nullptr, nullptr, nullptr, btab, remb,
          N_C, JPAD_C, INNER_C, INNER_C, INNER_C, JPAD_C,
          (long)N_C * INNER_C, 0, (long)N_C * JPAD_C);
      // fused top-8 threshold + masked softmax -> bf16 probs (zero-padded to JPAD)
      topk_softmax_k<<<grid1d((long)HEADS_C * N_C, 8), BLK, 0, stream>>>(
          dots, attn, HEADS_C * N_C);
      // attn @ v: per head z, out[b,i,z*64+d] = attn[z] @ vT[b,z]^T
      gemm_wmma<8, 1, 1, 0, true><<<dim3(1, N_C / 128, HEADS_C), BLK, 0, stream>>>(
          attn, vT + (size_t)b * INNER_C * JPAD_C,
          nullptr, aout + (size_t)b * N_C * INNER_C,
          nullptr, nullptr, nullptr, nullptr,
          N_C, DHEAD_C, JPAD_C, JPAD_C, JPAD_C, INNER_C,
          (long)N_C * JPAD_C, (long)DHEAD_C * JPAD_C, DHEAD_C);
    }
    // out-proj + bias + residual -> x_cur
    gemm_wmma<4, 2, 2, 2, false><<<dim3(4, 32, 1), BLK, 0, stream>>>(
        aout, woT, x_cur, nullptr, bo + (long)l * DIM_C, x_cur, nullptr, nullptr,
        (int)nTok, DIM_C, INNER_C, INNER_C, INNER_C, DIM_C, 0, 0, 0);

    // ---- feed-forward ----
    layernorm_k<<<grid1d(nTok, 8), BLK, 0, stream>>>(x_cur, ln2_g + l * DIM_C,
                                                     ln2_b + l * DIM_C, h_bf, (int)nTok);
    gemm_wmma<4, 2, 2, 3, true><<<dim3(16, 32, 1), BLK, 0, stream>>>(
        h_bf, w1T, nullptr, ffb, b1 + (long)l * FF_C, nullptr, nullptr, nullptr,
        (int)nTok, FF_C, DIM_C, DIM_C, DIM_C, FF_C, 0, 0, 0);
    gemm_wmma<4, 2, 2, 2, false><<<dim3(4, 32, 1), BLK, 0, stream>>>(
        ffb, w2T, x_cur, nullptr, b2 + (long)l * DIM_C, x_cur, nullptr, nullptr,
        (int)nTok, DIM_C, FF_C, FF_C, FF_C, DIM_C, 0, 0, 0);
  }

  copy_f32_k<<<grid1d(nTok * DIM_C, BLK), BLK, 0, stream>>>(x_cur, (float*)d_out,
                                                            nTok * DIM_C);
}
